// SSMElmanCell_29497835388999
// MI455X (gfx1250) — compile-verified, matrix operand order checked by reading, not run on hardware
//
#include <hip/hip_runtime.h>
#include <math.h>

// ---------------------------------------------------------------------------
// SSM Elman cell for MI455X (gfx1250, wave32, WMMA, TDM).
//   Prep   : f32->bf16 conversions of W_h and h0 (state mirror).
//   Phase 1: u[t,b,e]  = softplus(x·W_dt^T + b_dt) * (x·W_B^T)   (WMMA bf16)
//   Phase 2: h[t+1]    = sigmoid(a_log)*h[t] + h[t]·W_h^T + u[t] (persistent
//            WMMA scan, 8 WGs, TDM tensor_load_to_lds staging, grid barrier)
//   Phase 3: y[t]      = (x·W_C^T) * h[t+1]                      (WMMA bf16)
// ---------------------------------------------------------------------------

typedef __attribute__((ext_vector_type(16))) __bf16 v16bf;
typedef __attribute__((ext_vector_type(8)))  float  v8f;

#define T_DIM 1024
#define B_DIM 16
#define D_DIM 1024
#define TBROWS (T_DIM * B_DIM)   // 16384 flattened (t,b) rows

#define SCAN_WGS 8               // persistent scan workgroups
#define SCAN_THREADS 256         // 8 waves -> 128 output columns per WG
#define WSL_STRIDE 1028          // padded LDS row stride (ushorts): 1024 + 2x
                                 // 1-dword TDM pad per 256-dword interval

union FragAB { v16bf v; unsigned int d[8]; };
union FragC  { v8f   v; float        f[8]; };

__device__ __forceinline__ unsigned short f2bf(float f) {
  unsigned int x = __float_as_uint(f);
  x += 0x7FFFu + ((x >> 16) & 1u);
  return (unsigned short)(x >> 16);
}

// Build a 16x32 bf16 WMMA A/B fragment from an LDS tile laid out row-major
// [row][k] with `stride` ushorts per row (CDNA5 16-bit operand layout:
// lanes 0-15 = rows with K 0-7/16-23, lanes 16-31 = rows with K 8-15/24-31).
__device__ __forceinline__ v16bf load_frag(const unsigned short* base, int row0,
                                           int stride, int k0, int lane) {
  const int lrow  = lane & 15;
  const int khalf = lane >> 4;
  const unsigned short* p = base + (row0 + lrow) * stride + k0 + 8 * khalf;
  FragAB f;
#pragma unroll
  for (int i = 0; i < 4; ++i) {
    f.d[i]     = *(const unsigned int*)(p + 2 * i);
    f.d[i + 4] = *(const unsigned int*)(p + 16 + 2 * i);
  }
  return f.v;
}

// ---------------------------------------------------------------------------
// Tensor Data Mover: 2-D bf16 tile (tile_dim0 = 1024 contiguous K elements x
// `rows` rows) global -> LDS, with LDS padding of 1 dword per 256 dwords so
// the LDS row stride becomes 1028 ushorts (bank-conflict free fragments).
// D# built per CDNA5 ISA 8.3-8.5.  This toolchain exposes the 6-argument
// builtin: (u32x4 g0, i32x8 g1, i32x4 g2, i32x4 g3, i32x8 g4, i32 cpol).
// ---------------------------------------------------------------------------
#if __has_builtin(__builtin_amdgcn_tensor_load_to_lds)
#define HAVE_TDM 1
typedef __attribute__((ext_vector_type(4))) unsigned int u32x4;
typedef __attribute__((ext_vector_type(8))) int          i32x8;
typedef __attribute__((ext_vector_type(4))) int          i32x4;

__device__ __forceinline__ void tdm_load_2d_bf16(const void* gsrc,
                                                 unsigned lds_addr,
                                                 unsigned rows,
                                                 unsigned tensor_rows) {
  const unsigned long long ga = (unsigned long long)gsrc;
  u32x4 g0;
  g0[0] = 1u;                                        // count=1, user descriptor
  g0[1] = lds_addr;                                  // LDS byte address
  g0[2] = (unsigned)ga;                              // global_addr[31:0]
  g0[3] = (unsigned)((ga >> 32) & 0x01FFFFFFu)       // global_addr[56:32]
          | 0x80000000u;                             // type=2 ("image")
  i32x8 g1;
  g1[0] = (int)((1u << 16)      // data_size = 2 bytes
                | (1u << 20)    // pad_enable
                | (7u << 22));  // pad_interval = 256 dwords (pad_amount=1 dw)
  g1[1] = (int)(1024u << 16);   // tensor_dim0 = 1024 [bits 79:48 -> w1 hi]
  g1[2] = (int)(tensor_rows << 16);  // tensor_dim1[15:0]
  g1[3] = (int)(1024u << 16);   // tile_dim0 = 1024
  g1[4] = (int)rows;            // tile_dim1 = rows, tile_dim2 = 0
  g1[5] = 1024;                 // tensor_dim0_stride = 1024 elements
  g1[6] = 0;
  g1[7] = 0;
  const i32x4 z4 = {0, 0, 0, 0};
  const i32x8 z8 = {0, 0, 0, 0, 0, 0, 0, 0};
  __builtin_amdgcn_tensor_load_to_lds(g0, g1, z4, z4, z8, 0);
}

__device__ __forceinline__ void tdm_wait() {
  __builtin_amdgcn_s_wait_tensorcnt(0);
}
#else
#define HAVE_TDM 0
#endif

// Fallback/manual staging of a bf16 tile into the padded LDS layout.
__device__ __forceinline__ void stage_bf16_tile(unsigned short* dst,
                                                const unsigned short* src,
                                                int rows, int tid, int nthr) {
  for (int i = tid; i < rows * 512; i += nthr) {   // dword-granular copy
    const int n = i >> 9;
    const int k = (i & 511) * 2;
    const unsigned v = ((const unsigned int*)src)[(size_t)n * 512 + (i & 511)];
    *(unsigned int*)&dst[n * WSL_STRIDE + k + (k >= 512 ? 2 : 0)] = v;
  }
}

// ---------------------------------------------------------------------------
// Prep: elementwise f32 -> bf16.
// ---------------------------------------------------------------------------
__global__ __launch_bounds__(256) void cvt_bf16(const float* __restrict__ src,
                                                unsigned short* __restrict__ dst,
                                                int n) {
  const int i = blockIdx.x * 256 + threadIdx.x;
  if (i < n) dst[i] = f2bf(src[i]);
}

// ---------------------------------------------------------------------------
// Phase 1: dual-accumulator GEMM.  WG tile 128(M) x 64(N), K-step 32.
// ---------------------------------------------------------------------------
__global__ __launch_bounds__(256) void ssm_proj_u(
    const float* __restrict__ x, const float* __restrict__ Wdt,
    const float* __restrict__ Wb, const float* __restrict__ bdt,
    float* __restrict__ u) {
  __shared__ unsigned short xs[128 * 34];
  __shared__ unsigned short wdts[64 * 34];
  __shared__ unsigned short wbs[64 * 34];

  const int tid    = threadIdx.x;
  const int lane   = tid & 31;
  const int wave   = tid >> 5;
  const int n_base = blockIdx.x * 64;
  const int m_base = blockIdx.y * 128;

  FragC acc_dt[4], acc_b[4];
#pragma unroll
  for (int nt = 0; nt < 4; ++nt)
#pragma unroll
    for (int r = 0; r < 8; ++r) { acc_dt[nt].f[r] = 0.0f; acc_b[nt].f[r] = 0.0f; }

  for (int k0 = 0; k0 < D_DIM; k0 += 32) {
    for (int i = tid; i < 128 * 32; i += 256) {
      const int m = i >> 5, k = i & 31;
      xs[m * 34 + k] = f2bf(x[(size_t)(m_base + m) * D_DIM + k0 + k]);
    }
    for (int i = tid; i < 64 * 32; i += 256) {
      const int n = i >> 5, k = i & 31;
      wdts[n * 34 + k] = f2bf(Wdt[(size_t)(n_base + n) * D_DIM + k0 + k]);
      wbs[n * 34 + k]  = f2bf(Wb[(size_t)(n_base + n) * D_DIM + k0 + k]);
    }
    __syncthreads();

    if (k0 + 32 < D_DIM)
      __builtin_prefetch(&x[(size_t)(m_base + (tid >> 1)) * D_DIM + k0 + 32 +
                            (tid & 1) * 16], 0, 0);

    const v16bf a = load_frag(xs, wave * 16, 34, 0, lane);
#pragma unroll
    for (int nt = 0; nt < 4; ++nt) {
      const v16bf b0 = load_frag(wdts, nt * 16, 34, 0, lane);
      acc_dt[nt].v = __builtin_amdgcn_wmma_f32_16x16x32_bf16(
          false, a, false, b0, (short)0, acc_dt[nt].v, false, false);
      const v16bf b1 = load_frag(wbs, nt * 16, 34, 0, lane);
      acc_b[nt].v = __builtin_amdgcn_wmma_f32_16x16x32_bf16(
          false, a, false, b1, (short)0, acc_b[nt].v, false, false);
    }
    __syncthreads();
  }

  const int col   = lane & 15;
  const int khalf = lane >> 4;
#pragma unroll
  for (int nt = 0; nt < 4; ++nt) {
    const int C = n_base + nt * 16 + col;
    const float bd = bdt[C];
#pragma unroll
    for (int r = 0; r < 8; ++r) {
      const int R = m_base + wave * 16 + r + 8 * khalf;
      const float z  = acc_dt[nt].f[r] + bd;
      const float sp = (z > 15.0f) ? z : log1pf(expf(z));
      u[(size_t)R * D_DIM + C] = sp * acc_b[nt].f[r];
    }
  }
}

// ---------------------------------------------------------------------------
// Phase 2: persistent scan. 8 WGs x 8 waves, WG g owns columns [128g,128g+128).
// bf16 W_h slice (128x1024, ~257KB) + h[t] tile (16x1024, ~33KB) live in LDS
// (CDNA5 320KB WGP LDS). Staging uses the Tensor Data Mover when available.
// ---------------------------------------------------------------------------
__global__ __launch_bounds__(SCAN_THREADS) void ssm_scan(
    const unsigned short* __restrict__ whbf, const float* __restrict__ a_log,
    const float* __restrict__ u, float* __restrict__ hbuf,
    unsigned short* __restrict__ hbfA, unsigned short* __restrict__ hbfB,
    unsigned int* __restrict__ bar) {
  extern __shared__ unsigned short smem[];
  unsigned short* wsl = smem;                       // [128][WSL_STRIDE]
  unsigned short* hs  = smem + 128 * WSL_STRIDE;    // [16][WSL_STRIDE]

  const int tid   = threadIdx.x;
  const int lane  = tid & 31;
  const int wave  = tid >> 5;
  const int wg    = blockIdx.x;
  const int col   = lane & 15;
  const int khalf = lane >> 4;
  const int C     = wg * 128 + wave * 16 + col;     // this lane's column

  const float asig = 1.0f / (1.0f + expf(-a_log[C]));

  // ---- one-time: W_h slice (rows [128*wg, 128*wg+128)) into LDS ----
#if HAVE_TDM
  if (wave == 0) {
    tdm_load_2d_bf16(whbf + (size_t)wg * 128 * D_DIM,
                     (unsigned)(unsigned long long)(void*)wsl, 128u, 1024u);
    tdm_wait();
  }
#else
  stage_bf16_tile(wsl, whbf + (size_t)wg * 128 * D_DIM, 128, tid, SCAN_THREADS);
#endif
  __syncthreads();

  for (int t = 0; t < T_DIM; ++t) {
    const unsigned short* hsrc = (t & 1) ? hbfB : hbfA;  // bf16 h[t]
    unsigned short*       hdst = (t & 1) ? hbfA : hbfB;  // bf16 h[t+1]

    // ---- stage h[t] (16x1024 bf16, 32KB) into LDS ----
#if HAVE_TDM
    if (wave == 0) {
      tdm_load_2d_bf16(hsrc, (unsigned)(unsigned long long)(void*)hs, 16u, 16u);
      tdm_wait();
    }
#else
    stage_bf16_tile(hs, hsrc, 16, tid, SCAN_THREADS);
#endif
    __syncthreads();

    // ---- prefetch epilogue operands so they overlap the WMMA K-loop ----
    const float* hrow = hbuf + (size_t)t * B_DIM * D_DIM;      // f32 h[t]
    const float* ut   = u + (size_t)t * B_DIM * D_DIM;
    float hpv[8], uv[8];
#pragma unroll
    for (int r = 0; r < 8; ++r) {
      const int row = r + 8 * khalf;
      hpv[r] = hrow[row * D_DIM + C];
      uv[r]  = ut[row * D_DIM + C];
    }

    // ---- h[t] @ W_h^T for this wave's 16 columns ----
    FragC acc;
#pragma unroll
    for (int r = 0; r < 8; ++r) acc.f[r] = 0.0f;
    for (int kk = 0; kk < D_DIM; kk += 32) {
      const int kadj = kk + (kk >= 512 ? 2 : 0);   // TDM pad adjustment
      const v16bf a = load_frag(hs, 0, WSL_STRIDE, kadj, lane);
      const v16bf b = load_frag(wsl, wave * 16, WSL_STRIDE, kadj, lane);
      acc.v = __builtin_amdgcn_wmma_f32_16x16x32_bf16(
          false, a, false, b, (short)0, acc.v, false, false);
    }

    // ---- h[t+1] = A*h[t] + h@W_h^T + u[t]; f32 to output, bf16 mirror ----
    float* hnext = hbuf + (size_t)(t + 1) * B_DIM * D_DIM;
#pragma unroll
    for (int r = 0; r < 8; ++r) {
      const int row = r + 8 * khalf;
      const float hv = asig * hpv[r] + acc.f[r] + uv[r];
      hnext[row * D_DIM + C] = hv;
      hdst[row * D_DIM + C]  = f2bf(hv);
    }

    // ---- device-wide barrier: monotone counter, target = 8*(t+1) ----
    __threadfence();
    __syncthreads();
    if (tid == 0) {
      __hip_atomic_fetch_add(bar, 1u, __ATOMIC_ACQ_REL, __HIP_MEMORY_SCOPE_AGENT);
      const unsigned int target = (unsigned int)SCAN_WGS * (unsigned int)(t + 1);
      while (__hip_atomic_load(bar, __ATOMIC_ACQUIRE, __HIP_MEMORY_SCOPE_AGENT) < target)
        __builtin_amdgcn_s_sleep(2);
    }
    __syncthreads();
    __threadfence();
  }
}

// ---------------------------------------------------------------------------
// Phase 3: y[t] = (x @ W_C^T) * h[t+1]; hbuf row = x row + 16.
// ---------------------------------------------------------------------------
__global__ __launch_bounds__(256) void ssm_proj_y(
    const float* __restrict__ x, const float* __restrict__ Wc,
    const float* __restrict__ hbuf, float* __restrict__ y) {
  __shared__ unsigned short xs[128 * 34];
  __shared__ unsigned short wcs[64 * 34];

  const int tid    = threadIdx.x;
  const int lane   = tid & 31;
  const int wave   = tid >> 5;
  const int n_base = blockIdx.x * 64;
  const int m_base = blockIdx.y * 128;

  FragC acc[4];
#pragma unroll
  for (int nt = 0; nt < 4; ++nt)
#pragma unroll
    for (int r = 0; r < 8; ++r) acc[nt].f[r] = 0.0f;

  for (int k0 = 0; k0 < D_DIM; k0 += 32) {
    for (int i = tid; i < 128 * 32; i += 256) {
      const int m = i >> 5, k = i & 31;
      xs[m * 34 + k] = f2bf(x[(size_t)(m_base + m) * D_DIM + k0 + k]);
    }
    for (int i = tid; i < 64 * 32; i += 256) {
      const int n = i >> 5, k = i & 31;
      wcs[n * 34 + k] = f2bf(Wc[(size_t)(n_base + n) * D_DIM + k0 + k]);
    }
    __syncthreads();

    if (k0 + 32 < D_DIM)
      __builtin_prefetch(&x[(size_t)(m_base + (tid >> 1)) * D_DIM + k0 + 32 +
                            (tid & 1) * 16], 0, 0);

    const v16bf a = load_frag(xs, wave * 16, 34, 0, lane);
#pragma unroll
    for (int nt = 0; nt < 4; ++nt) {
      const v16bf b = load_frag(wcs, nt * 16, 34, 0, lane);
      acc[nt].v = __builtin_amdgcn_wmma_f32_16x16x32_bf16(
          false, a, false, b, (short)0, acc[nt].v, false, false);
    }
    __syncthreads();
  }

  const int col   = lane & 15;
  const int khalf = lane >> 4;
#pragma unroll
  for (int nt = 0; nt < 4; ++nt) {
    const int C = n_base + nt * 16 + col;
#pragma unroll
    for (int r = 0; r < 8; ++r) {
      const int R = m_base + wave * 16 + r + 8 * khalf;
      y[(size_t)R * D_DIM + C] =
          acc[nt].f[r] * hbuf[(size_t)(R + 16) * D_DIM + C];
    }
  }
}

// ---------------------------------------------------------------------------
// Host-side launch.  d_in: x, h0, W_B, W_C, W_dt, W_h, a_log, b_dt.
// d_out = y[T,B,D] ++ h[T+1,B,D].
// ws: [0,256) barrier | [256, +64MB) u | whbf 2MB | hbfA 32KB | hbfB 32KB.
// ---------------------------------------------------------------------------
extern "C" void kernel_launch(void* const* d_in, const int* in_sizes, int n_in,
                              void* d_out, int out_size, void* d_ws, size_t ws_size,
                              hipStream_t stream) {
  (void)in_sizes; (void)n_in; (void)out_size; (void)ws_size;
  const float* x     = (const float*)d_in[0];
  const float* h0    = (const float*)d_in[1];
  const float* W_B   = (const float*)d_in[2];
  const float* W_C   = (const float*)d_in[3];
  const float* W_dt  = (const float*)d_in[4];
  const float* W_h   = (const float*)d_in[5];
  const float* a_log = (const float*)d_in[6];
  const float* b_dt  = (const float*)d_in[7];

  float* y    = (float*)d_out;
  float* hbuf = y + (size_t)T_DIM * B_DIM * D_DIM;   // h, [(T+1)*16][1024]

  char* ws = (char*)d_ws;
  unsigned int*   bar  = (unsigned int*)ws;
  float*          u    = (float*)(ws + 256);
  unsigned short* whbf = (unsigned short*)(ws + 256 + (size_t)TBROWS * D_DIM * 4);
  unsigned short* hbfA = whbf + (size_t)D_DIM * D_DIM;
  unsigned short* hbfB = hbfA + (size_t)B_DIM * D_DIM;

  (void)hipMemsetAsync(d_ws, 0, 256, stream);        // reset grid barrier
  (void)hipMemcpyAsync(hbuf, h0, (size_t)B_DIM * D_DIM * sizeof(float),
                       hipMemcpyDeviceToDevice, stream);  // h[0] = h0 (f32)

  // prep: bf16 W_h and bf16 h0 mirror
  cvt_bf16<<<(D_DIM * D_DIM + 255) / 256, 256, 0, stream>>>(W_h, whbf,
                                                            D_DIM * D_DIM);
  cvt_bf16<<<(B_DIM * D_DIM + 255) / 256, 256, 0, stream>>>(h0, hbfA,
                                                            B_DIM * D_DIM);

  ssm_proj_u<<<dim3(D_DIM / 64, TBROWS / 128), 256, 0, stream>>>(
      x, W_dt, W_B, b_dt, u);

  const size_t smem = (size_t)(128 * WSL_STRIDE + 16 * WSL_STRIDE) *
                      sizeof(unsigned short);        // ~290 KB of 320 KB LDS
  ssm_scan<<<dim3(SCAN_WGS), SCAN_THREADS, smem, stream>>>(
      whbf, a_log, u, hbuf, hbfA, hbfB, bar);

  ssm_proj_y<<<dim3(D_DIM / 64, TBROWS / 128), 256, 0, stream>>>(
      x, W_C, hbuf, y);
}
